// Repro_17609365913846
// MI455X (gfx1250) — matrix operand order repaired and checked by reading, not compiled, not run on hardware
//
#include <hip/hip_runtime.h>
#include <hip/hip_bf16.h>

#define BB 8
#define SS 1024
#define DD 768
#define HH 12
#define DHH 64
#define FF 3072
#define EMB_SCALE 27.712812921102035f
#define ATTN_SCALE 0.125f

typedef _Float16 f16;
typedef _Float16 v8h  __attribute__((ext_vector_type(8)));
typedef _Float16 v16h __attribute__((ext_vector_type(16)));
typedef float    v8f  __attribute__((ext_vector_type(8)));

// ---------------------------------------------------------------------------
// WMMA fragment loaders (layouts per CDNA5 ISA §7.12.2)
// A (16x32 f16): lane = 16g + r holds row r; elems 0-7 = K[8g..8g+7],
//                elems 8-15 = K[16+8g..16+8g+7]  -> two 16B loads.
// B (32x16 f16): lane = 16g + n holds column n; elems 0-7 = K[8g..8g+7],
//                elems 8-15 = K[16+8g..16+8g+7] -> two 16B loads (data stored as B^T rows).
// ---------------------------------------------------------------------------
__device__ __forceinline__ v16h load_fragA(const f16* base, int ld, int lane) {
  const int r = lane & 15, g = lane >> 4;
  const f16* p = base + (size_t)r * ld + 8 * g;
  v8h lo = *(const v8h*)p;
  v8h hi = *(const v8h*)(p + 16);
  return __builtin_shufflevector(lo, hi, 0,1,2,3,4,5,6,7,8,9,10,11,12,13,14,15);
}

__device__ __forceinline__ v16h load_fragB(const f16* base, int ld, int lane) {
  const int n = lane & 15, g = lane >> 4;
  const f16* p = base + (size_t)n * ld + 8 * g;
  v8h lo = *(const v8h*)p;
  v8h hi = *(const v8h*)(p + 16);
  return __builtin_shufflevector(lo, hi, 0,1,2,3,4,5,6,7,8,9,10,11,12,13,14,15);
}

__device__ __forceinline__ v8f wmma16(v16h a, v16h b, v8f c) {
  return __builtin_amdgcn_wmma_f32_16x16x32_f16(false, a, false, b, (short)0, c,
                                                false, false);
}

// ---------------------------------------------------------------------------
// f32 -> f16 conversion (weights)
// ---------------------------------------------------------------------------
__global__ void cvt_f32_f16(const float* __restrict__ in, f16* __restrict__ out,
                            int n) {
  int i = blockIdx.x * blockDim.x + threadIdx.x;
  int stride = gridDim.x * blockDim.x;
  for (; i < n; i += stride) out[i] = (f16)in[i];
}

// ---------------------------------------------------------------------------
// Embedding + LayerNorm1: one block (256 thr) per token, 3 elems/thread
// ---------------------------------------------------------------------------
__global__ void embed_ln1(const int* __restrict__ ids,
                          const float* __restrict__ tok,
                          const float* __restrict__ pos,
                          const float* __restrict__ gam,
                          const float* __restrict__ bet,
                          float* __restrict__ hF, f16* __restrict__ hH) {
  const int t = blockIdx.x;            // token 0..B*S-1
  const int s = t & (SS - 1);
  const int tid = threadIdx.x;
  const int id = ids[t];
  const float* te = tok + (size_t)id * DD;
  const float* pe = pos + (size_t)(s + 2) * DD;
  float x[3];
  float lsum = 0.f, lsq = 0.f;
#pragma unroll
  for (int e = 0; e < 3; ++e) {
    int d = tid + 256 * e;
    float v = te[d] * EMB_SCALE + pe[d];
    x[e] = v; lsum += v; lsq += v * v;
  }
  __shared__ float r1[256], r2[256];
  r1[tid] = lsum; r2[tid] = lsq;
  __syncthreads();
  for (int off = 128; off > 0; off >>= 1) {
    if (tid < off) { r1[tid] += r1[tid + off]; r2[tid] += r2[tid + off]; }
    __syncthreads();
  }
  float mean = r1[0] * (1.0f / DD);
  float var  = r2[0] * (1.0f / DD) - mean * mean;
  float rstd = rsqrtf(var + 1e-5f);
  float* ho = hF + (size_t)t * DD;
  f16*   hh = hH + (size_t)t * DD;
#pragma unroll
  for (int e = 0; e < 3; ++e) {
    int d = tid + 256 * e;
    float hv = (x[e] - mean) * rstd * gam[d] + bet[d];
    ho[d] = hv;
    hh[d] = (f16)hv;
  }
}

// ---------------------------------------------------------------------------
// LayerNorm2: reads f32 activations, emits f32 (to d_out) + f16
// ---------------------------------------------------------------------------
__global__ void ln2_kernel(const float* __restrict__ a,
                           const float* __restrict__ gam,
                           const float* __restrict__ bet,
                           float* __restrict__ h2F, f16* __restrict__ h2H) {
  const int t = blockIdx.x;
  const int tid = threadIdx.x;
  const float* ap = a + (size_t)t * DD;
  float x[3];
  float lsum = 0.f, lsq = 0.f;
#pragma unroll
  for (int e = 0; e < 3; ++e) {
    int d = tid + 256 * e;
    float v = ap[d];
    x[e] = v; lsum += v; lsq += v * v;
  }
  __shared__ float r1[256], r2[256];
  r1[tid] = lsum; r2[tid] = lsq;
  __syncthreads();
  for (int off = 128; off > 0; off >>= 1) {
    if (tid < off) { r1[tid] += r1[tid + off]; r2[tid] += r2[tid + off]; }
    __syncthreads();
  }
  float mean = r1[0] * (1.0f / DD);
  float var  = r2[0] * (1.0f / DD) - mean * mean;
  float rstd = rsqrtf(var + 1e-5f);
  float* ho = h2F + (size_t)t * DD;
  f16*   hh = h2H + (size_t)t * DD;
#pragma unroll
  for (int e = 0; e < 3; ++e) {
    int d = tid + 256 * e;
    float hv = (x[e] - mean) * rstd * gam[d] + bet[d];
    ho[d] = hv;
    hh[d] = (f16)hv;
  }
}

// ---------------------------------------------------------------------------
// Generic WMMA GEMM: C[M,N] = A[M,K] @ W[N,K]^T + bias, with epilogue modes.
// Block: 256 threads = 8 waves, block tile 64(M) x 256(N), wave tile 32x64:
// 12 b128 loads per 8 WMMAs. Working set fits MI455X 192MB L2 -> direct
// global loads (no LDS staging needed).
// ---------------------------------------------------------------------------
#define GM_NONE 0
#define GM_VT   1   // store f16 transposed as V^T[b,h,dh,s]
#define GM_GELU 2   // gelu(erf) then f16 store
#define GM_RES  3   // f32 store with residual add

__global__ void __launch_bounds__(256)
gemm_wmma(const f16* __restrict__ A, const f16* __restrict__ W,
          const float* __restrict__ bias, int M, int N, int K, float scale,
          int mode, f16* __restrict__ outH, const float* __restrict__ resF,
          float* __restrict__ outF) {
  const int lane = threadIdx.x & 31;
  const int wave = threadIdx.x >> 5;
  const int wr = wave >> 2;          // 0..1 (M)
  const int wc = wave & 3;           // 0..3 (N)
  const int m0 = blockIdx.y * 64 + wr * 32;
  const int n0 = blockIdx.x * 256 + wc * 64;

  v8f acc[2][4] = {};
  for (int k0 = 0; k0 < K; k0 += 32) {
    v16h a0 = load_fragA(A + (size_t)m0 * K + k0, K, lane);
    v16h a1 = load_fragA(A + (size_t)(m0 + 16) * K + k0, K, lane);
    v16h b0 = load_fragB(W + (size_t)(n0 +  0) * K + k0, K, lane);
    v16h b1 = load_fragB(W + (size_t)(n0 + 16) * K + k0, K, lane);
    v16h b2 = load_fragB(W + (size_t)(n0 + 32) * K + k0, K, lane);
    v16h b3 = load_fragB(W + (size_t)(n0 + 48) * K + k0, K, lane);
    acc[0][0] = wmma16(a0, b0, acc[0][0]);
    acc[0][1] = wmma16(a0, b1, acc[0][1]);
    acc[0][2] = wmma16(a0, b2, acc[0][2]);
    acc[0][3] = wmma16(a0, b3, acc[0][3]);
    acc[1][0] = wmma16(a1, b0, acc[1][0]);
    acc[1][1] = wmma16(a1, b1, acc[1][1]);
    acc[1][2] = wmma16(a1, b2, acc[1][2]);
    acc[1][3] = wmma16(a1, b3, acc[1][3]);
  }

  const int cn = lane & 15;
  const int rg = (lane >> 4) * 8;
#pragma unroll
  for (int i = 0; i < 2; ++i) {
#pragma unroll
    for (int j = 0; j < 4; ++j) {
      const int n = n0 + j * 16 + cn;
      const float bv = bias[n];
#pragma unroll
      for (int e = 0; e < 8; ++e) {
        const int m = m0 + i * 16 + rg + e;
        float v = (acc[i][j][e] + bv) * scale;
        if (mode == GM_GELU) {
          v = 0.5f * v * (erff(v * 0.70710678118654752f) + 1.0f);
          outH[(size_t)m * N + n] = (f16)v;
        } else if (mode == GM_RES) {
          outF[(size_t)m * N + n] = resF[(size_t)m * N + n] + v;
        } else if (mode == GM_VT) {
          const int bb = m >> 10, sI = m & (SS - 1);
          const int hI = n >> 6, dI = n & 63;
          outH[(((size_t)bb * HH + hI) * DHH + dI) * SS + sI] = (f16)v;
        } else {
          outH[(size_t)m * N + n] = (f16)v;
        }
      }
    }
  }
}

// ---------------------------------------------------------------------------
// Flash attention: one wave per 16 q-rows; 32 keys / iteration.
// Q pre-scaled by 1/sqrt(dh). V supplied transposed [b,h,dh,s] so B-fragments
// for P@V are contiguous. P transposed via 1KB per-wave LDS buffer.
// ---------------------------------------------------------------------------
__global__ void __launch_bounds__(128)
attn_wmma(const f16* __restrict__ Q, const f16* __restrict__ Kc,
          const f16* __restrict__ Vt, f16* __restrict__ O) {
  const int lane = threadIdx.x & 31;
  const int w = threadIdx.x >> 5;
  const int gw = blockIdx.x * 4 + w;
  const int qt = gw & 63;
  const int bh = gw >> 6;
  const int b = bh / HH;
  const int h = bh % HH;
  const int qbase = qt * 16;
  const int r = lane & 15;
  const int g = lane >> 4;

  __shared__ __align__(16) f16 plds[4][16 * 32];
  f16* pw = plds[w];

  const f16* qp = Q + ((size_t)(b * SS + qbase) * DD) + h * DHH;
  const v16h qa0 = load_fragA(qp, DD, lane);        // dh 0..31
  const v16h qa1 = load_fragA(qp + 32, DD, lane);   // dh 32..63

  float rm[8], rl[8];
#pragma unroll
  for (int j = 0; j < 8; ++j) { rm[j] = -__builtin_inff(); rl[j] = 0.f; }
  v8f accO[4] = {};

  const f16* kbp = Kc + ((size_t)b * SS) * DD + h * DHH;
  const f16* vtb = Vt + ((size_t)(b * HH + h) * DHH) * SS;
  const int nIter = (qbase + 47) >> 5;   // key tiles of 32 covering causal span

  for (int it = 0; it < nIter; ++it) {
    const int kb = it * 32;
    const f16* kp0 = kbp + (size_t)kb * DD;
    const f16* kp1 = kbp + (size_t)(kb + 16) * DD;
    v16h k00 = load_fragB(kp0, DD, lane);
    v16h k01 = load_fragB(kp0 + 32, DD, lane);
    v16h k10 = load_fragB(kp1, DD, lane);
    v16h k11 = load_fragB(kp1 + 32, DD, lane);
    v8f s0 = {}; s0 = wmma16(qa0, k00, s0); s0 = wmma16(qa1, k01, s0);
    v8f s1 = {}; s1 = wmma16(qa0, k10, s1); s1 = wmma16(qa1, k11, s1);

    float corr[8];
#pragma unroll
    for (int j = 0; j < 8; ++j) {
      const int mAbs = qbase + g * 8 + j;
      const int key0 = kb + r;
      float v0 = (key0 <= mAbs) ? s0[j] : -__builtin_inff();
      float v1 = (key0 + 16 <= mAbs) ? s1[j] : -__builtin_inff();
      float mx = fmaxf(v0, v1);
#pragma unroll
      for (int off = 1; off < 16; off <<= 1)
        mx = fmaxf(mx, __shfl_xor(mx, off, 32));
      const float nm = fmaxf(rm[j], mx);
      const float c = __expf(rm[j] - nm);
      const float p0 = __expf(v0 - nm);
      const float p1 = __expf(v1 - nm);
      float ps = p0 + p1;
#pragma unroll
      for (int off = 1; off < 16; off <<= 1) ps += __shfl_xor(ps, off, 32);
      rl[j] = rl[j] * c + ps;
      rm[j] = nm;
      corr[j] = c;
      pw[(g * 8 + j) * 32 + r] = (f16)p0;
      pw[(g * 8 + j) * 32 + 16 + r] = (f16)p1;
    }
#pragma unroll
    for (int c4 = 0; c4 < 4; ++c4)
#pragma unroll
      for (int j = 0; j < 8; ++j) accO[c4][j] *= corr[j];

    asm volatile("s_wait_dscnt 0" ::: "memory");
    const v16h pa = load_fragA(pw, 32, lane);   // P tile 16x32 as A

#pragma unroll
    for (int c4 = 0; c4 < 4; ++c4) {
      v16h vb = load_fragB(vtb + (size_t)(16 * c4) * SS + kb, SS, lane);
      accO[c4] = wmma16(pa, vb, accO[c4]);
    }
  }

  f16* op = O + ((size_t)(b * SS + qbase) * DD) + h * DHH;
#pragma unroll
  for (int j = 0; j < 8; ++j) {
    const float inv = 1.0f / rl[j];
#pragma unroll
    for (int c4 = 0; c4 < 4; ++c4)
      op[(size_t)(g * 8 + j) * DD + 16 * c4 + r] = (f16)(accO[c4][j] * inv);
  }
}

// ---------------------------------------------------------------------------
extern "C" void kernel_launch(void* const* d_in, const int* in_sizes, int n_in,
                              void* d_out, int out_size, void* d_ws,
                              size_t ws_size, hipStream_t stream) {
  const int*   ids  = (const int*)d_in[0];
  const float* tok  = (const float*)d_in[1];
  const float* pos  = (const float*)d_in[2];
  const float* ln1w = (const float*)d_in[3];
  const float* ln1b = (const float*)d_in[4];
  const float* wq   = (const float*)d_in[5];
  const float* bq   = (const float*)d_in[6];
  const float* wk   = (const float*)d_in[7];
  const float* bk   = (const float*)d_in[8];
  const float* wv   = (const float*)d_in[9];
  const float* bv   = (const float*)d_in[10];
  const float* wo   = (const float*)d_in[11];
  const float* bo   = (const float*)d_in[12];
  const float* ln2w = (const float*)d_in[13];
  const float* ln2b = (const float*)d_in[14];
  const float* w1   = (const float*)d_in[15];
  const float* b1   = (const float*)d_in[16];
  const float* w2   = (const float*)d_in[17];
  const float* b2   = (const float*)d_in[18];
  float* out = (float*)d_out;

  char* ws = (char*)d_ws;
  size_t off = 0;
  auto alloc = [&](size_t bytes) {
    void* p = ws + off;
    off += (bytes + 255) & ~(size_t)255;
    return p;
  };
  const size_t NT = (size_t)BB * SS;              // 8192 tokens
  float* hF  = (float*)alloc(NT * DD * 4);        // x -> h(LN1) -> a (in place)
  f16* hH    = (f16*)alloc(NT * DD * 2);
  f16* q16   = (f16*)alloc(NT * DD * 2);
  f16* k16   = (f16*)alloc(NT * DD * 2);
  f16* vT16  = (f16*)alloc(NT * DD * 2);          // [b,h,dh,s]
  f16* o16   = (f16*)alloc(NT * DD * 2);
  f16* h2H   = (f16*)alloc(NT * DD * 2);
  f16* g16   = (f16*)alloc(NT * FF * 2);
  f16* wq16  = (f16*)alloc((size_t)DD * DD * 2);
  f16* wk16  = (f16*)alloc((size_t)DD * DD * 2);
  f16* wv16  = (f16*)alloc((size_t)DD * DD * 2);
  f16* wo16  = (f16*)alloc((size_t)DD * DD * 2);
  f16* w116  = (f16*)alloc((size_t)FF * DD * 2);
  f16* w216  = (f16*)alloc((size_t)DD * FF * 2);

  // 1) weights -> f16
  cvt_f32_f16<<<1024, 256, 0, stream>>>(wq, wq16, DD * DD);
  cvt_f32_f16<<<1024, 256, 0, stream>>>(wk, wk16, DD * DD);
  cvt_f32_f16<<<1024, 256, 0, stream>>>(wv, wv16, DD * DD);
  cvt_f32_f16<<<1024, 256, 0, stream>>>(wo, wo16, DD * DD);
  cvt_f32_f16<<<2048, 256, 0, stream>>>(w1, w116, FF * DD);
  cvt_f32_f16<<<2048, 256, 0, stream>>>(w2, w216, DD * FF);

  // 2) embedding + LN1
  embed_ln1<<<(int)NT, 256, 0, stream>>>(ids, tok, pos, ln1w, ln1b, hF, hH);

  // 3) QKV projections (block tile 64x256)
  dim3 gD(DD / 256, (int)(NT / 64));
  gemm_wmma<<<gD, 256, 0, stream>>>(hH, wq16, bq, (int)NT, DD, DD, ATTN_SCALE,
                                    GM_NONE, q16, nullptr, nullptr);
  gemm_wmma<<<gD, 256, 0, stream>>>(hH, wk16, bk, (int)NT, DD, DD, 1.0f,
                                    GM_NONE, k16, nullptr, nullptr);
  gemm_wmma<<<gD, 256, 0, stream>>>(hH, wv16, bv, (int)NT, DD, DD, 1.0f,
                                    GM_VT, vT16, nullptr, nullptr);

  // 4) causal flash attention (B*H*(S/16) = 6144 waves, 4 waves/block)
  attn_wmma<<<1536, 128, 0, stream>>>(q16, k16, vT16, o16);

  // 5) output projection + residual (a = h + o@wo^T + bo), in place into hF
  gemm_wmma<<<gD, 256, 0, stream>>>(o16, wo16, bo, (int)NT, DD, DD, 1.0f,
                                    GM_RES, nullptr, hF, hF);

  // 6) LN2 -> d_out (f32) + f16 copy
  ln2_kernel<<<(int)NT, 256, 0, stream>>>(hF, ln2w, ln2b, out, h2H);

  // 7) FFN1 + gelu
  dim3 gF(FF / 256, (int)(NT / 64));
  gemm_wmma<<<gF, 256, 0, stream>>>(h2H, w116, b1, (int)NT, FF, DD, 1.0f,
                                    GM_GELU, g16, nullptr, nullptr);

  // 8) FFN2 + residual (out = h2 + ff), in place into d_out
  gemm_wmma<<<gD, 256, 0, stream>>>(g16, w216, b2, (int)NT, DD, FF, 1.0f,
                                    GM_RES, nullptr, out, out);
}
